// GraphTransformerEncoder_23768349016099
// MI455X (gfx1250) — compile-verified, matrix operand order checked by reading, not run on hardware
//
#include <hip/hip_runtime.h>
#include <hip/hip_bf16.h>
#include <math.h>

// ---------------------------------------------------------------------------
// GraphTransformerEncoder for MI455X (gfx1250, wave32, WMMA).
// Dense GEMMs use v_wmma_f32_16x16x32_f16 (f32->f16 operands, f32 accum) with
// LDS double-buffering: tile kt+1 is loaded (global_load_b128) into registers
// while WMMAs run on tile kt, then written to the alternate LDS buffer; one
// barrier per K-step. Node-dim buffers padded to the 128-row tile so the GEMM
// is bounds-check free. Graph scatter parts are f32 VALU + global float
// atomics; We (16x768, 48KB) staged in LDS per block.
// ---------------------------------------------------------------------------

typedef __attribute__((ext_vector_type(16))) _Float16 v16h;
typedef __attribute__((ext_vector_type(8)))  float    v8f;

#define NN    10000
#define MPAD  10112          // 79 * 128, padded node count for GEMM tiles
#define EE    100000
#define DD    768
#define HH    4
#define CCH   192
#define EDIMF 16
#define NRELS 10
#define LAYERS 3

// ------------------------- helpers -----------------------------------------

__device__ __forceinline__ unsigned int f2mono(float f) {
  unsigned int u = __float_as_uint(f);
  return (u & 0x80000000u) ? ~u : (u | 0x80000000u);
}
__device__ __forceinline__ float mono2f(unsigned int u) {
  return (u & 0x80000000u) ? __uint_as_float(u & 0x7fffffffu)
                           : __uint_as_float(~u);
}

// ------------------------- tiny elementwise kernels -------------------------

__global__ void copy_f(const float* __restrict__ a, float* __restrict__ b, int n) {
  int i = blockIdx.x * blockDim.x + threadIdx.x;
  if (i < n) b[i] = a[i];
}

__global__ void zero_f(float* __restrict__ p, int n) {
  int i = blockIdx.x * blockDim.x + threadIdx.x;
  if (i < n) p[i] = 0.0f;
}

__global__ void init_softmax_state(unsigned int* __restrict__ m,
                                   float* __restrict__ den, int n) {
  int i = blockIdx.x * blockDim.x + threadIdx.x;
  if (i < n) { m[i] = f2mono(-INFINITY); den[i] = 0.0f; }
}

__global__ void gelu_k(float* __restrict__ p, int n) {
  int i = blockIdx.x * blockDim.x + threadIdx.x;
  if (i < n) {
    float x = p[i];
    p[i] = 0.5f * x * (1.0f + erff(x * 0.7071067811865475f));
  }
}

// ------------------------- edge feature projection ---------------------------
// ef[e] = concat(rel_emb[rid], w) @ W_edge + b_edge   -> [E,16]

__global__ void edge_feat_k(const float* __restrict__ eattr,
                            const float* __restrict__ rel_emb,
                            const float* __restrict__ W_edge,
                            const float* __restrict__ b_edge,
                            float* __restrict__ ef) {
  int idx = blockIdx.x * blockDim.x + threadIdx.x;
  if (idx >= EE * EDIMF) return;
  int e = idx >> 4, j = idx & 15;
  int rid = (int)eattr[e * 2 + 0];
  rid = rid < 0 ? 0 : (rid > NRELS - 1 ? NRELS - 1 : rid);
  float w = eattr[e * 2 + 1];
  float s = b_edge[j];
#pragma unroll
  for (int i = 0; i < EDIMF - 1; ++i)
    s += rel_emb[rid * (EDIMF - 1) + i] * W_edge[i * EDIMF + j];
  s += w * W_edge[(EDIMF - 1) * EDIMF + j];
  ef[idx] = s;
}

// ------------------------- WMMA GEMM: C = A(MxK) * B(KxN) + bias -------------
// M padded (MPAD): no bounds checks. Block tile 128x64, BK=32, 8 waves (4x2),
// each wave 32x32 via 2x2 v_wmma_f32_16x16x32_f16 accumulators.
// Double-buffered LDS; one __syncthreads per K-step.
// Fragment layouts per CDNA5 ISA 7.12.2:
//   A 16x32 f16: lane L -> row M=L&15; hi=L>>4; VGPR pair p holds
//     K = (p<4?0:16) + (hi?8:0) + 2*(p&3) + {0,1}   (contiguous runs of 8
//     halfs -> ds_load_b128 merging)
//   B staged K-transposed in LDS so lanes see the same contiguous K runs.
//   C/D v8f: lane L -> col N=L&15, VGPR r -> row M = r + (hi?8:0)

__global__ __launch_bounds__(256)
void gemm_wmma_f16(const float* __restrict__ A, const float* __restrict__ B,
                   const float* __restrict__ bias, float* __restrict__ Cout,
                   int K, int Nn) {
  constexpr int BM = 128, BN = 64, BK = 32, LDW = 40;  // 40 = BK + 8 pad
  __shared__ _Float16 As[2][BM][LDW];
  __shared__ _Float16 Bs[2][BN][LDW];

  const int tid  = threadIdx.x;
  const int lane = tid & 31;
  const int wave = tid >> 5;
  const int wr   = wave >> 1;   // 0..3
  const int wc   = wave & 1;    // 0..1
  const int hi   = lane >> 4;
  const int l16  = lane & 15;
  const int rowBase = blockIdx.x * BM;
  const int colBase = blockIdx.y * BN;

  // K-invariant staging coordinates: A = 1024 float4 (4/thread),
  // B = 512 float4 (2/thread).
  const float* aP[4];
  int aR[4], aC[4];
#pragma unroll
  for (int u = 0; u < 4; ++u) {
    int i = tid + u * 256;
    aR[u] = i >> 3;            // 8 float4 per 32-wide row
    aC[u] = (i & 7) << 2;
    aP[u] = A + (size_t)(rowBase + aR[u]) * K + aC[u];
  }
  const float* bP[2];
  int bKr[2], bNc[2];
#pragma unroll
  for (int u = 0; u < 2; ++u) {
    int i = tid + u * 256;
    bKr[u] = i >> 4;           // 16 float4 per 64-wide row
    bNc[u] = (i & 15) << 2;
    bP[u] = B + (size_t)bKr[u] * Nn + colBase + bNc[u];
  }

  const int NT = K / BK;

  // stage tile 0 into buffer 0
  float4 av[4], bv4[2];
#pragma unroll
  for (int u = 0; u < 4; ++u) av[u] = *reinterpret_cast<const float4*>(aP[u]);
#pragma unroll
  for (int u = 0; u < 2; ++u) bv4[u] = *reinterpret_cast<const float4*>(bP[u]);
#pragma unroll
  for (int u = 0; u < 4; ++u) {
    As[0][aR[u]][aC[u] + 0] = (_Float16)av[u].x;
    As[0][aR[u]][aC[u] + 1] = (_Float16)av[u].y;
    As[0][aR[u]][aC[u] + 2] = (_Float16)av[u].z;
    As[0][aR[u]][aC[u] + 3] = (_Float16)av[u].w;
  }
#pragma unroll
  for (int u = 0; u < 2; ++u) {
    Bs[0][bNc[u] + 0][bKr[u]] = (_Float16)bv4[u].x;
    Bs[0][bNc[u] + 1][bKr[u]] = (_Float16)bv4[u].y;
    Bs[0][bNc[u] + 2][bKr[u]] = (_Float16)bv4[u].z;
    Bs[0][bNc[u] + 3][bKr[u]] = (_Float16)bv4[u].w;
  }

  const v8f vzero = {0.f, 0.f, 0.f, 0.f, 0.f, 0.f, 0.f, 0.f};
  v8f acc[2][2];
  acc[0][0] = vzero; acc[0][1] = vzero; acc[1][0] = vzero; acc[1][1] = vzero;

  for (int kt = 0; kt < NT; ++kt) {
    const int buf = kt & 1;
    __syncthreads();  // tile[buf] staged; buf^1 free to overwrite

    const bool more = (kt + 1) < NT;
    if (more) {
      // issue next tile's global loads now; they overlap the WMMAs below
#pragma unroll
      for (int u = 0; u < 4; ++u) {
        aP[u] += BK;
        av[u] = *reinterpret_cast<const float4*>(aP[u]);
      }
#pragma unroll
      for (int u = 0; u < 2; ++u) {
        bP[u] += (size_t)BK * Nn;
        bv4[u] = *reinterpret_cast<const float4*>(bP[u]);
      }
      if (kt + 2 < NT) {
        __builtin_prefetch(aP[0] + BK, 0, 1);
        __builtin_prefetch(bP[0] + (size_t)BK * Nn, 0, 1);
      }
    }

    // build fragments from tile[buf] and multiply
    v16h af[2], bfg[2];
#pragma unroll
    for (int t = 0; t < 2; ++t) {
      const _Float16* ar = &As[buf][wr * 32 + t * 16 + l16][0];
      const _Float16* br = &Bs[buf][wc * 32 + t * 16 + l16][0];
#pragma unroll
      for (int p = 0; p < 8; ++p) {
        int kk = ((p & 4) ? 16 : 0) + (hi ? 8 : 0) + ((p & 3) << 1);
        af[t][2 * p]      = ar[kk];
        af[t][2 * p + 1]  = ar[kk + 1];
        bfg[t][2 * p]     = br[kk];
        bfg[t][2 * p + 1] = br[kk + 1];
      }
    }
#pragma unroll
    for (int tm = 0; tm < 2; ++tm)
#pragma unroll
      for (int tn = 0; tn < 2; ++tn)
        acc[tm][tn] = __builtin_amdgcn_wmma_f32_16x16x32_f16(
            false, af[tm], false, bfg[tn], (short)0, acc[tm][tn], false, false);

    if (more) {
      const int nb = buf ^ 1;
#pragma unroll
      for (int u = 0; u < 4; ++u) {
        As[nb][aR[u]][aC[u] + 0] = (_Float16)av[u].x;
        As[nb][aR[u]][aC[u] + 1] = (_Float16)av[u].y;
        As[nb][aR[u]][aC[u] + 2] = (_Float16)av[u].z;
        As[nb][aR[u]][aC[u] + 3] = (_Float16)av[u].w;
      }
#pragma unroll
      for (int u = 0; u < 2; ++u) {
        Bs[nb][bNc[u] + 0][bKr[u]] = (_Float16)bv4[u].x;
        Bs[nb][bNc[u] + 1][bKr[u]] = (_Float16)bv4[u].y;
        Bs[nb][bNc[u] + 2][bKr[u]] = (_Float16)bv4[u].z;
        Bs[nb][bNc[u] + 3][bKr[u]] = (_Float16)bv4[u].w;
      }
    }
  }

  // epilogue: guard-free strided stores (row per accumulator register)
#pragma unroll
  for (int tm = 0; tm < 2; ++tm) {
#pragma unroll
    for (int tn = 0; tn < 2; ++tn) {
      int gcol = colBase + wc * 32 + tn * 16 + l16;
      float bvv = bias[gcol];
      size_t base = (size_t)(rowBase + wr * 32 + tm * 16 + (hi ? 8 : 0)) * Nn + gcol;
#pragma unroll
      for (int r = 0; r < 8; ++r)
        Cout[base + (size_t)r * Nn] = acc[tm][tn][r] + bvv;
    }
  }
}

// ------------------------- edge attention logits + segment max ---------------
// One wave per edge. alpha[e,h] = scale * sum_c q[dst,c]*(k[src,c]+e[e,c]).
// e[e,c] on the fly from ef (16 regs, 4x float4) and We staged in LDS.
// Channel chunks: c = lane + 32*i, i in 0..23; head = i/6 (192 = 6*32 aligns).

__global__ __launch_bounds__(256)
void edge_alpha_k(const float* __restrict__ qb, const float* __restrict__ kb,
                  const float* __restrict__ ef, const float* __restrict__ WeL,
                  const int* __restrict__ eidx, float* __restrict__ alpha,
                  unsigned int* __restrict__ mmax) {
  __shared__ float WeS[EDIMF * DD];
  for (int i = threadIdx.x; i < EDIMF * DD; i += 256) WeS[i] = WeL[i];
  __syncthreads();

  int e = blockIdx.x * 8 + (threadIdx.x >> 5);
  int lane = threadIdx.x & 31;
  if (e >= EE) return;
  int src = eidx[e], dst = eidx[EE + e];

  const float4* ef4 = reinterpret_cast<const float4*>(ef + (size_t)e * EDIMF);
  float4 f0 = ef4[0], f1 = ef4[1], f2 = ef4[2], f3 = ef4[3];
  float efv[EDIMF] = {f0.x, f0.y, f0.z, f0.w, f1.x, f1.y, f1.z, f1.w,
                      f2.x, f2.y, f2.z, f2.w, f3.x, f3.y, f3.z, f3.w};

  float part[HH] = {0.f, 0.f, 0.f, 0.f};
#pragma unroll
  for (int i = 0; i < 24; ++i) {
    int c = lane + (i << 5);
    float ec = 0.f;
#pragma unroll
    for (int j = 0; j < EDIMF; ++j) ec += efv[j] * WeS[j * DD + c];
    float kj = kb[(size_t)src * DD + c] + ec;
    part[i / 6] += qb[(size_t)dst * DD + c] * kj;
  }
  const float scale = 0.07216878364870323f;  // 1/sqrt(192)
#pragma unroll
  for (int h = 0; h < HH; ++h) {
    float s = part[h];
#pragma unroll
    for (int off = 16; off > 0; off >>= 1) s += __shfl_xor(s, off, 32);
    s *= scale;
    if (lane == h) {
      alpha[(size_t)e * HH + h] = s;
      atomicMax(&mmax[(size_t)dst * HH + h], f2mono(s));
    }
  }
}

// ------------------------- softmax exp + segment sum -------------------------

__global__ void softmax_exp_k(const int* __restrict__ eidx,
                              const unsigned int* __restrict__ mmax,
                              float* __restrict__ alpha,
                              float* __restrict__ denom) {
  int idx = blockIdx.x * blockDim.x + threadIdx.x;
  if (idx >= EE * HH) return;
  int e = idx >> 2, h = idx & 3;
  int dst = eidx[EE + e];
  float mval = mono2f(mmax[(size_t)dst * HH + h]);
  float a = expf(alpha[idx] - mval);
  alpha[idx] = a;
  atomicAdd(&denom[(size_t)dst * HH + h], a);
}

// ------------------------- weighted message scatter --------------------------
// hn[dst] += (v[src] + e) * a   (hn pre-initialized with skip projection)

__global__ __launch_bounds__(256)
void edge_message_k(const float* __restrict__ vb, const float* __restrict__ ef,
                    const float* __restrict__ WeL, const int* __restrict__ eidx,
                    const float* __restrict__ alpha,
                    const float* __restrict__ denom, float* __restrict__ hn) {
  __shared__ float WeS[EDIMF * DD];
  for (int i = threadIdx.x; i < EDIMF * DD; i += 256) WeS[i] = WeL[i];
  __syncthreads();

  int e = blockIdx.x * 8 + (threadIdx.x >> 5);
  int lane = threadIdx.x & 31;
  if (e >= EE) return;
  int src = eidx[e], dst = eidx[EE + e];

  const float4* ef4 = reinterpret_cast<const float4*>(ef + (size_t)e * EDIMF);
  float4 f0 = ef4[0], f1 = ef4[1], f2 = ef4[2], f3 = ef4[3];
  float efv[EDIMF] = {f0.x, f0.y, f0.z, f0.w, f1.x, f1.y, f1.z, f1.w,
                      f2.x, f2.y, f2.z, f2.w, f3.x, f3.y, f3.z, f3.w};

  float coef[HH];
#pragma unroll
  for (int h = 0; h < HH; ++h)
    coef[h] = alpha[(size_t)e * HH + h] /
              (denom[(size_t)dst * HH + h] + 1e-16f);

#pragma unroll
  for (int i = 0; i < 24; ++i) {
    int c = lane + (i << 5);
    float ec = 0.f;
#pragma unroll
    for (int j = 0; j < EDIMF; ++j) ec += efv[j] * WeS[j * DD + c];
    float msg = (vb[(size_t)src * DD + c] + ec) * coef[i / 6];
    atomicAdd(&hn[(size_t)dst * DD + c], msg);
  }
}

// ------------------------- LayerNorm + global mean pool ----------------------

__global__ __launch_bounds__(256)
void ln_meanpool_k(const float* __restrict__ h, const float* __restrict__ g,
                   const float* __restrict__ b, float* __restrict__ out) {
  int node = blockIdx.x;
  const float* row = h + (size_t)node * DD;
  int t = threadIdx.x;
  float x0 = row[t], x1 = row[t + 256], x2 = row[t + 512];
  float s = x0 + x1 + x2;
  float ss = x0 * x0 + x1 * x1 + x2 * x2;

  __shared__ float rs[8], rss[8];
  int lane = t & 31, wid = t >> 5;
#pragma unroll
  for (int off = 16; off > 0; off >>= 1) {
    s  += __shfl_xor(s, off, 32);
    ss += __shfl_xor(ss, off, 32);
  }
  if (lane == 0) { rs[wid] = s; rss[wid] = ss; }
  __syncthreads();
  if (t == 0) {
    float a = 0.f, a2 = 0.f;
#pragma unroll
    for (int w = 0; w < 8; ++w) { a += rs[w]; a2 += rss[w]; }
    rs[0] = a; rss[0] = a2;
  }
  __syncthreads();
  float mu = rs[0] * (1.0f / DD);
  float var = rss[0] * (1.0f / DD) - mu * mu;
  float inv = rsqrtf(var + 1e-5f);
  const float invN = 1.0f / (float)NN;
#pragma unroll
  for (int kch = 0; kch < 3; ++kch) {
    int c = t + kch * 256;
    float y = (row[c] - mu) * inv * g[c] + b[c];
    atomicAdd(&out[c], y * invN);
  }
}

// ------------------------- host orchestration --------------------------------

extern "C" void kernel_launch(void* const* d_in, const int* in_sizes, int n_in,
                              void* d_out, int out_size, void* d_ws, size_t ws_size,
                              hipStream_t stream) {
  (void)in_sizes; (void)n_in; (void)out_size; (void)ws_size;
  const float* x       = (const float*)d_in[0];
  const float* eattr   = (const float*)d_in[1];
  const int*   eidx    = (const int*)  d_in[2];
  const float* rel_emb = (const float*)d_in[3];
  const float* W_edge  = (const float*)d_in[4];
  const float* b_edge  = (const float*)d_in[5];
  const float* Wk      = (const float*)d_in[6];
  const float* bk      = (const float*)d_in[7];
  const float* Wq      = (const float*)d_in[8];
  const float* bq      = (const float*)d_in[9];
  const float* Wv      = (const float*)d_in[10];
  const float* bv      = (const float*)d_in[11];
  const float* We      = (const float*)d_in[12];
  const float* Wskip   = (const float*)d_in[13];
  const float* bskip   = (const float*)d_in[14];
  const float* ln_g    = (const float*)d_in[15];
  const float* ln_b    = (const float*)d_in[16];
  float* out = (float*)d_out;

  // workspace layout (floats); node-dim buffers padded to MPAD rows
  float* ws = (float*)d_ws;
  float* h     = ws;
  float* hn    = h  + (size_t)MPAD * DD;
  float* qb    = hn + (size_t)MPAD * DD;
  float* kb    = qb + (size_t)MPAD * DD;
  float* vb    = kb + (size_t)MPAD * DD;
  float* ef    = vb + (size_t)MPAD * DD;        // E*16
  float* alpha = ef + (size_t)EE * EDIMF;       // E*H
  unsigned int* mmax = (unsigned int*)(alpha + (size_t)EE * HH);  // N*H
  float* denom = (float*)(mmax + (size_t)NN * HH);                // N*H

  const int ND = NN * DD;
  dim3 gemmGrid(MPAD / 128, DD / 64);
  const int edgeBlocks = (EE + 7) / 8;

  // h = x (real rows only; padded rows are don't-care)
  copy_f<<<(ND + 255) / 256, 256, 0, stream>>>(x, h, ND);
  // edge features (layer-independent)
  edge_feat_k<<<(EE * EDIMF + 255) / 256, 256, 0, stream>>>(
      eattr, rel_emb, W_edge, b_edge, ef);

  float* cur = h;
  float* nxt = hn;
  for (int l = 0; l < LAYERS; ++l) {
    const float* WqL = Wq + (size_t)l * DD * DD;
    const float* WkL = Wk + (size_t)l * DD * DD;
    const float* WvL = Wv + (size_t)l * DD * DD;
    const float* WsL = Wskip + (size_t)l * DD * DD;
    const float* WeLl = We + (size_t)l * EDIMF * DD;

    gemm_wmma_f16<<<gemmGrid, 256, 0, stream>>>(cur, WqL, bq + l * DD, qb, DD, DD);
    gemm_wmma_f16<<<gemmGrid, 256, 0, stream>>>(cur, WkL, bk + l * DD, kb, DD, DD);
    gemm_wmma_f16<<<gemmGrid, 256, 0, stream>>>(cur, WvL, bv + l * DD, vb, DD, DD);
    // nxt = cur @ Wskip + bskip  (messages accumulate on top)
    gemm_wmma_f16<<<gemmGrid, 256, 0, stream>>>(cur, WsL, bskip + l * DD, nxt, DD, DD);

    init_softmax_state<<<(NN * HH + 255) / 256, 256, 0, stream>>>(mmax, denom, NN * HH);
    edge_alpha_k<<<edgeBlocks, 256, 0, stream>>>(qb, kb, ef, WeLl, eidx, alpha, mmax);
    softmax_exp_k<<<(EE * HH + 255) / 256, 256, 0, stream>>>(eidx, mmax, alpha, denom);
    edge_message_k<<<edgeBlocks, 256, 0, stream>>>(vb, ef, WeLl, eidx, alpha, denom, nxt);

    if (l < LAYERS - 1)
      gelu_k<<<(ND + 255) / 256, 256, 0, stream>>>(nxt, ND);

    float* tmp = cur; cur = nxt; nxt = tmp;
  }

  zero_f<<<(DD + 255) / 256, 256, 0, stream>>>(out, DD);
  ln_meanpool_k<<<NN, 256, 0, stream>>>(cur, ln_g, ln_b, out);
}
